// LSTMCell_10453950399020
// MI455X (gfx1250) — compile-verified, matrix operand order checked by reading, not run on hardware
//
#include <hip/hip_runtime.h>
#include <hip/hip_bf16.h>
#include <stdint.h>

typedef _Float16 v16h __attribute__((ext_vector_type(16)));
typedef _Float16 v8h  __attribute__((ext_vector_type(8)));
typedef _Float16 v4h  __attribute__((ext_vector_type(4)));
typedef float    v8f  __attribute__((ext_vector_type(8)));
typedef float    v4f  __attribute__((ext_vector_type(4)));
typedef unsigned int u32x4 __attribute__((ext_vector_type(4)));
typedef int          i32x8 __attribute__((ext_vector_type(8)));
typedef int          i32x4 __attribute__((ext_vector_type(4)));

#define BDIM   8192
#define HDIM   2048
#define KSRC   2048                    // IN == H == 2048
#define BK     32                      // K per WMMA step
#define BM     256                     // rows per workgroup
#define BNH    32                      // h-columns per workgroup
#define BNBIG  128                     // 4 gates * BNH
#define LSTR   40                      // padded LDS row stride (halves): 80B row pitch
#define A_HALVES (BM * LSTR)           // 10240
#define B_HALVES (BNBIG * LSTR)        // 5120
#define BUF_HALVES (A_HALVES + B_HALVES)
#define STEPS  128                     // 64 steps over x/Wx, 64 over h/Wh

#if __has_builtin(__builtin_amdgcn_tensor_load_to_lds) && __has_builtin(__builtin_amdgcn_s_wait_tensorcnt)
#define HAVE_TDM 1
#else
#define HAVE_TDM 0
#endif

// ---------------- fp32 -> fp16 pre-convert (one-time, into d_ws) ----------------
__global__ void cvt_f32_to_f16(const float* __restrict__ in,
                               _Float16* __restrict__ out, int n4)
{
    int i = blockIdx.x * blockDim.x + threadIdx.x;
    if (i < n4) {
        v4f v = *(const v4f*)(in + (size_t)i * 4);
        v4h o;
        o.x = (_Float16)v.x; o.y = (_Float16)v.y;
        o.z = (_Float16)v.z; o.w = (_Float16)v.w;
        *(v4h*)(out + (size_t)i * 4) = o;
    }
}

#if HAVE_TDM
// Issue one TDM tile load: (tile0 x tile1 x tile2) elements of 2 bytes, row stride
// s0 elems, plane stride s1 elems, LDS padding of 4 DWORDs every 16 DWORDs
// (=> 80B LDS row pitch for a 64B tile row, matching LSTR=40 halves).
// 6-arg builtin: VADDR0..VADDR3 descriptor groups, VADDR4 unused (zero), cpol.
__device__ __forceinline__ void tdm_tile_load(unsigned lds_byte, unsigned long long gaddr,
                                              unsigned tile0, unsigned tile1, unsigned tile2,
                                              unsigned long long s0, unsigned long long s1)
{
    u32x4 g0;
    g0.x = 1u;                                            // count=1, user descriptor
    g0.y = lds_byte;                                      // LDS byte address
    g0.z = (unsigned)(gaddr & 0xFFFFFFFFull);             // global_addr[31:0]
    g0.w = (unsigned)((gaddr >> 32) & 0x1FFFFFFull) | (2u << 30);  // addr[56:32] | type=2

    const unsigned dim0 = 2048, dim1 = 1u << 20;          // ample (no OOB clipping needed)
    i32x8 g1;
    g1.s0 = (int)((1u << 16) | (1u << 20) | (3u << 22) | (3u << 25)); // data_size=2B, pad_en, int=16dw, amt=4dw
    g1.s1 = (int)((dim0 & 0xFFFFu) << 16);                // tensor_dim0[15:0]
    g1.s2 = (int)((dim0 >> 16) | ((dim1 & 0xFFFFu) << 16));
    g1.s3 = (int)((dim1 >> 16) | (tile0 << 16));          // tensor_dim1[31:16] | tile_dim0
    g1.s4 = (int)(tile1 | (tile2 << 16));                 // tile_dim1 | tile_dim2
    g1.s5 = (int)(s0 & 0xFFFFFFFFull);                    // tensor_dim0_stride[31:0]
    g1.s6 = (int)((unsigned)(s0 >> 32) | ((unsigned)(s1 & 0xFFFFull) << 16));
    g1.s7 = (int)((s1 >> 16) & 0xFFFFFFFFull);            // tensor_dim1_stride[47:16]

    i32x4 g2; g2.x = 64; g2.y = 0; g2.z = 0; g2.w = 0;    // tensor_dim2 ample, no dim3
    i32x4 g3; g3.x = 0;  g3.y = 0; g3.z = 0; g3.w = 0;
    i32x8 g4 = {0, 0, 0, 0, 0, 0, 0, 0};                  // VADDR4: unused, zero-filled

    __builtin_amdgcn_tensor_load_to_lds(g0, g1, g2, g3, g4, 0);
}
#endif

// Manual fallback fill (global -> VGPR -> LDS), 512 threads
__device__ __forceinline__ void fill_tile(_Float16* dst, int tid, int row0, int hcol0,
                                          const _Float16* __restrict__ A,
                                          const _Float16* __restrict__ W, int k0)
{
#pragma unroll
    for (int i = 0; i < 2; ++i) {
        int c = tid + i * 512;
        int r = c >> 2, s = c & 3;
        v8h v = *(const v8h*)(A + (size_t)(row0 + r) * KSRC + k0 + s * 8);
        *(v8h*)(dst + r * LSTR + s * 8) = v;
    }
    {
        int n = tid >> 2, s = tid & 3;
        int g = n >> 5, nl = n & 31;
        v8h v = *(const v8h*)(W + ((size_t)g * HDIM + (size_t)(hcol0 + nl)) * KSRC + k0 + s * 8);
        *(v8h*)(dst + A_HALVES + n * LSTR + s * 8) = v;
    }
}

// Per-lane 16-bit WMMA fragment: lane (l, hi) holds K in [hi*8, hi*8+8) U [16+hi*8, 16+hi*8+8)
__device__ __forceinline__ v16h frag_from_lds(const _Float16* base, int row, int hi)
{
    v8h lo = *(const v8h*)(base + row * LSTR + hi * 8);
    v8h up = *(const v8h*)(base + row * LSTR + 16 + hi * 8);
    return __builtin_shufflevector(lo, up, 0,1,2,3,4,5,6,7,8,9,10,11,12,13,14,15);
}

__device__ __forceinline__ float fast_sigmoid(float x)
{
    return __builtin_amdgcn_rcpf(1.f + __expf(-x));
}
__device__ __forceinline__ float fast_tanh(float x)
{
    // exact at +/-inf: exp(2x)->inf => 1; exp(2x)->0 => -1
    return 1.f - 2.f * __builtin_amdgcn_rcpf(1.f + __expf(2.f * x));
}

// ---------------- fused LSTM GEMM + epilogue ----------------
__global__ __launch_bounds__(512)
void lstm_fused_wmma(const _Float16* __restrict__ X16,
                     const _Float16* __restrict__ Hs16,
                     const _Float16* __restrict__ WX16,  // [4][2048][2048] f,i,o,c
                     const _Float16* __restrict__ WH16,  // [4][2048][2048] f,i,o,c
                     const float* __restrict__ c_in,
                     const float* __restrict__ bf_p, const float* __restrict__ bi_p,
                     const float* __restrict__ bo_p, const float* __restrict__ bc_p,
                     float* __restrict__ h_out, float* __restrict__ c_out)
{
    __shared__ _Float16 lds[2 * BUF_HALVES];   // 60 KB of 320 KB WGP LDS

    const int tid   = threadIdx.x;
    const int wave  = tid >> 5;
    const int lane  = tid & 31;
    const int l     = lane & 15;
    const int hi    = lane >> 4;
    const int waveM = wave & 7;      // 8 wave-rows  -> 32 rows each
    const int waveN = wave >> 3;     // 2 wave-cols  -> 16 h-cols each
    const int row0  = blockIdx.y * BM;
    const int hcol0 = blockIdx.x * BNH;

    const v8f vzero = {0.f, 0.f, 0.f, 0.f, 0.f, 0.f, 0.f, 0.f};
    v8f acc[4][2];
#pragma unroll
    for (int g = 0; g < 4; ++g) { acc[g][0] = vzero; acc[g][1] = vzero; }

#if HAVE_TDM
    // ---- TDM-fed double-buffered pipeline: wave 0 drives the Tensor Data Mover ----
    const unsigned ldsBase = (unsigned)(uintptr_t)&lds[0];
    if (wave == 0) {
        tdm_tile_load(ldsBase, (unsigned long long)(uintptr_t)(X16 + (size_t)row0 * KSRC),
                      BK, BM, 1, KSRC, 0);
        tdm_tile_load(ldsBase + A_HALVES * 2,
                      (unsigned long long)(uintptr_t)(WX16 + (size_t)hcol0 * KSRC),
                      BK, BNH, 4, KSRC, (unsigned long long)HDIM * KSRC);
        __builtin_amdgcn_s_wait_tensorcnt(0);
    }
    __syncthreads();

    int buf = 0;
    for (int s = 0; s < STEPS; ++s) {
        if (wave == 0 && s + 1 < STEPS) {
            int ns = s + 1;
            const _Float16* A = (ns & 64) ? Hs16 : X16;
            const _Float16* W = (ns & 64) ? WH16 : WX16;
            int k0 = (ns & 63) * BK;
            unsigned dstA = ldsBase + (unsigned)((buf ^ 1) * BUF_HALVES * 2);
            tdm_tile_load(dstA, (unsigned long long)(uintptr_t)(A + (size_t)row0 * KSRC + k0),
                          BK, BM, 1, KSRC, 0);
            tdm_tile_load(dstA + A_HALVES * 2,
                          (unsigned long long)(uintptr_t)(W + (size_t)hcol0 * KSRC + k0),
                          BK, BNH, 4, KSRC, (unsigned long long)HDIM * KSRC);
        }
        const _Float16* As = lds + buf * BUF_HALVES;
        const _Float16* Bs = As + A_HALVES;

        v16h a0 = frag_from_lds(As, waveM * 32 + l, hi);
        v16h a1 = frag_from_lds(As, waveM * 32 + 16 + l, hi);
#pragma unroll
        for (int g = 0; g < 4; ++g) {
            v16h b = frag_from_lds(Bs, g * 32 + waveN * 16 + l, hi);
            acc[g][0] = __builtin_amdgcn_wmma_f32_16x16x32_f16(
                            false, a0, false, b, (short)0, acc[g][0], false, false);
            acc[g][1] = __builtin_amdgcn_wmma_f32_16x16x32_f16(
                            false, a1, false, b, (short)0, acc[g][1], false, false);
        }
        if (wave == 0) __builtin_amdgcn_s_wait_tensorcnt(0);
        __syncthreads();
        buf ^= 1;
    }
#else
    // ---- fallback: per-thread global loads + ds_store double buffering ----
    fill_tile(lds, tid, row0, hcol0, X16, WX16, 0);
    __syncthreads();

    int buf = 0;
    for (int s = 0; s < STEPS; ++s) {
        if (s + 1 < STEPS) {
            int ns = s + 1;
            const _Float16* A = (ns & 64) ? Hs16 : X16;
            const _Float16* W = (ns & 64) ? WH16 : WX16;
            fill_tile(lds + (buf ^ 1) * BUF_HALVES, tid, row0, hcol0, A, W, (ns & 63) * BK);
        }
        const _Float16* As = lds + buf * BUF_HALVES;
        const _Float16* Bs = As + A_HALVES;

        v16h a0 = frag_from_lds(As, waveM * 32 + l, hi);
        v16h a1 = frag_from_lds(As, waveM * 32 + 16 + l, hi);
#pragma unroll
        for (int g = 0; g < 4; ++g) {
            v16h b = frag_from_lds(Bs, g * 32 + waveN * 16 + l, hi);
            acc[g][0] = __builtin_amdgcn_wmma_f32_16x16x32_f16(
                            false, a0, false, b, (short)0, acc[g][0], false, false);
            acc[g][1] = __builtin_amdgcn_wmma_f32_16x16x32_f16(
                            false, a1, false, b, (short)0, acc[g][1], false, false);
        }
        __syncthreads();
        buf ^= 1;
    }
#endif

    // ---- epilogue: gates -> (h_new, c_new), all four gates local to this wave ----
    const int hcol = hcol0 + waveN * 16 + l;
    const float bf = bf_p[hcol], bi = bi_p[hcol], bo = bo_p[hcol], bc = bc_p[hcol];

#pragma unroll
    for (int mi = 0; mi < 2; ++mi) {
#pragma unroll
        for (int r = 0; r < 8; ++r) {
            const int row = row0 + waveM * 32 + mi * 16 + r + 8 * hi;  // C/D layout: M = r + 8*hi
            float gf = acc[0][mi][r] + bf;
            float gi = acc[1][mi][r] + bi;
            float go = acc[2][mi][r] + bo;
            float gc = acc[3][mi][r] + bc;
            float f  = fast_sigmoid(gf);
            float i  = fast_sigmoid(gi);
            float o  = fast_sigmoid(go);
            float ct = fast_tanh(gc);
            float cold = c_in[(size_t)row * HDIM + hcol];
            float cn = f * cold + i * ct;
            float hn = o * fast_tanh(cn);
            h_out[(size_t)row * HDIM + hcol] = hn;
            c_out[(size_t)row * HDIM + hcol] = cn;
        }
    }
}

// ---------------- host launcher ----------------
extern "C" void kernel_launch(void* const* d_in, const int* in_sizes, int n_in,
                              void* d_out, int out_size, void* d_ws, size_t ws_size,
                              hipStream_t stream)
{
    // d_in order: x, h, c, w_xf, w_hf, b_f, w_xi, w_hi, b_i, w_xo, w_ho, b_o, w_xc, w_hc, b_c
    _Float16* ws   = (_Float16*)d_ws;
    _Float16* X16  = ws;                                   // 8192*2048
    _Float16* Hs16 = ws + (size_t)16777216;                // 8192*2048
    _Float16* WX16 = ws + (size_t)33554432;                // 4 * 2048*2048
    _Float16* WH16 = ws + (size_t)50331648;                // 4 * 2048*2048  (total 128 MiB)

    const size_t GSZ = (size_t)HDIM * KSRC;                // 4194304 per gate matrix

    auto conv = [&](int idx, _Float16* dst, size_t n) {
        int n4 = (int)(n / 4);
        cvt_f32_to_f16<<<(n4 + 255) / 256, 256, 0, stream>>>((const float*)d_in[idx], dst, n4);
    };

    conv(0,  X16,              (size_t)BDIM * KSRC);
    conv(1,  Hs16,             (size_t)BDIM * KSRC);
    conv(3,  WX16 + 0 * GSZ,   GSZ);   // w_xf
    conv(6,  WX16 + 1 * GSZ,   GSZ);   // w_xi
    conv(9,  WX16 + 2 * GSZ,   GSZ);   // w_xo
    conv(12, WX16 + 3 * GSZ,   GSZ);   // w_xc
    conv(4,  WH16 + 0 * GSZ,   GSZ);   // w_hf
    conv(7,  WH16 + 1 * GSZ,   GSZ);   // w_hi
    conv(10, WH16 + 2 * GSZ,   GSZ);   // w_ho
    conv(13, WH16 + 3 * GSZ,   GSZ);   // w_hc

    float* h_out = (float*)d_out;
    float* c_out = (float*)d_out + (size_t)BDIM * HDIM;

    dim3 grid(HDIM / BNH, BDIM / BM);   // (64, 32)
    lstm_fused_wmma<<<grid, 512, 0, stream>>>(
        X16, Hs16, WX16, WH16,
        (const float*)d_in[2],
        (const float*)d_in[5], (const float*)d_in[8],
        (const float*)d_in[11], (const float*)d_in[14],
        h_out, c_out);
}
